// ELMo_75995151335884
// MI455X (gfx1250) — matrix operand order repaired and checked by reading, not compiled, not run on hardware
//
#include <hip/hip_runtime.h>
#include <math.h>

typedef __attribute__((ext_vector_type(16))) _Float16 v16h;
typedef __attribute__((ext_vector_type(8)))  float    v8f;

#define CHAR_EMBC 16
#define NFILT     1400
#define NFILTP    1408          // padded filter count (88 tiles of 16)
#define CONVKP    64            // padded conv K (48 -> 64)
#define NPOS      14
#define WORDLEN   16
#define SEQLEN    128
#define BATCHN    32
#define NWORDS    (SEQLEN * BATCHN)   // 4096 (row n = s*32 + b)
#define DDIM      512
#define HDIM      256
#define VOCABN    50000

// ---------------- f16 fragment loaders (pure b128 loads, no conversion) ----------------

// A-fragment body: two contiguous 8-half (16B) runs at p and p+16.
__device__ __forceinline__ v16h fragA(const _Float16* p) {
  union { v16h v; float4 q[2]; } u;
  u.q[0] = *(const float4*)(p);         // K = kb .. kb+7
  u.q[1] = *(const float4*)(p + 16);    // K = kb+16 .. kb+23
  return u.v;
}
// B-fragment body: two contiguous 8-half (16B) runs at p and p+16.
__device__ __forceinline__ v16h fragB(const _Float16* p) {
  union { v16h v; float4 q[2]; } u;
  u.q[0] = *(const float4*)(p);
  u.q[1] = *(const float4*)(p + 16);
  return u.v;
}

// Pointer-forming wrappers (used by charcnn / lstm_scan; gemm uses induction pointers).
__device__ __forceinline__ v16h loadA_h(const _Float16* A, int lda, int m0, int k0, int lane) {
  return fragA(A + (long)(m0 + (lane & 15)) * lda + (k0 + ((lane >> 4) << 3)));
}
__device__ __forceinline__ v16h loadB_h(const _Float16* W, int ldw, int n0, int k0, int lane) {
  return fragB(W + (long)(n0 + (lane & 15)) * ldw + (k0 + ((lane >> 4) << 3)));
}

// ---------------- one-time f32 -> f16 repack with zero padding ----------------
__global__ void pack_pad_h(const float* __restrict__ src, _Float16* __restrict__ dst,
                           int N, int K, int Npad, int Kpad) {
  long idx = (long)blockIdx.x * blockDim.x + threadIdx.x;
  if (idx >= (long)Npad * Kpad) return;
  int n = (int)(idx / Kpad), k = (int)(idx - (long)n * Kpad);
  dst[idx] = (_Float16)((n < N && k < K) ? src[(long)n * K + k] : 0.f);
}

// ---------------- blocked GEMM: C[M,N] = A[M,K] @ W[N,K]^T + b1 + b2 ----------------
// f16 operands, f32 accumulate. One wave computes a 64x16 strip (4 M-tiles share B),
// explicit double buffering + induction-variable pointers (no per-step addr recompute).
// Requires M%64==0, N%16==0, K%32==0. Cf (f32) and/or Ch (f16) outputs, either nullable.
__global__ void __launch_bounds__(256, 1)
gemm_wt(const _Float16* __restrict__ A, const _Float16* __restrict__ W,
        const float* __restrict__ b1, const float* __restrict__ b2,
        float* __restrict__ Cf, _Float16* __restrict__ Ch,
        int M, int N, int K) {
  int lane    = threadIdx.x & 31;
  int ntiles  = N >> 4;
  int mblocks = M >> 6;
  int task    = blockIdx.x * (blockDim.x >> 5) + (threadIdx.x >> 5);
  if (task >= mblocks * ntiles) return;          // wave-uniform
  int mb = task / ntiles, nt = task - mb * ntiles;
  int m0 = mb << 6, n0 = nt << 4;
  int l15 = lane & 15, hi = lane >> 4;

  // lane-resolved stream pointers, advanced by 32 halves per k-step
  const _Float16* pB  = W + (long)(n0 + l15) * K + (hi << 3);
  const _Float16* pA0 = A + (long)(m0 + l15) * K + (hi << 3);
  const _Float16* pA1 = pA0 + (long)16 * K;
  const _Float16* pA2 = pA0 + (long)32 * K;
  const _Float16* pA3 = pA0 + (long)48 * K;

  v8f acc0 = {}, acc1 = {}, acc2 = {}, acc3 = {};

  // prologue: fragments for step 0
  v16h b  = fragB(pB);
  v16h a0 = fragA(pA0);
  v16h a1 = fragA(pA1);
  v16h a2 = fragA(pA2);
  v16h a3 = fragA(pA3);

  int steps = K >> 5;
  for (int it = 1; it < steps; ++it) {
    pB += 32; pA0 += 32; pA1 += 32; pA2 += 32; pA3 += 32;
    __builtin_prefetch(pB + 32, 0, 1);           // speculative: safe past end
    // issue next step's loads into the alternate fragment set
    v16h bn  = fragB(pB);
    v16h an0 = fragA(pA0);
    v16h an1 = fragA(pA1);
    v16h an2 = fragA(pA2);
    v16h an3 = fragA(pA3);
    // consume previous step's fragments (loads already landed / overlapping)
    acc0 = __builtin_amdgcn_wmma_f32_16x16x32_f16(false, a0, false, b, (short)0, acc0, false, false);
    acc1 = __builtin_amdgcn_wmma_f32_16x16x32_f16(false, a1, false, b, (short)0, acc1, false, false);
    acc2 = __builtin_amdgcn_wmma_f32_16x16x32_f16(false, a2, false, b, (short)0, acc2, false, false);
    acc3 = __builtin_amdgcn_wmma_f32_16x16x32_f16(false, a3, false, b, (short)0, acc3, false, false);
    b = bn; a0 = an0; a1 = an1; a2 = an2; a3 = an3;
  }
  // epilogue: last step
  acc0 = __builtin_amdgcn_wmma_f32_16x16x32_f16(false, a0, false, b, (short)0, acc0, false, false);
  acc1 = __builtin_amdgcn_wmma_f32_16x16x32_f16(false, a1, false, b, (short)0, acc1, false, false);
  acc2 = __builtin_amdgcn_wmma_f32_16x16x32_f16(false, a2, false, b, (short)0, acc2, false, false);
  acc3 = __builtin_amdgcn_wmma_f32_16x16x32_f16(false, a3, false, b, (short)0, acc3, false, false);

  int n = n0 + l15;
  float bias = 0.f;
  if (b1) bias += b1[n];
  if (b2) bias += b2[n];
  int rbase = hi << 3;
  v8f accs[4] = {acc0, acc1, acc2, acc3};
  #pragma unroll
  for (int t = 0; t < 4; ++t) {
    long base = (long)(m0 + (t << 4) + rbase) * N + n;
    #pragma unroll
    for (int r = 0; r < 8; ++r) {
      float v = accs[t][r] + bias;
      if (Cf) Cf[base + (long)r * N] = v;
      if (Ch) Ch[base + (long)r * N] = (_Float16)v;
    }
  }
}

// ---------------- fused char-embed + conv(K=48) + relu + max-over-pos ----------------
// One block per word: f16 im2col tile staged in LDS, 8 waves sweep 88 filter tiles.
__global__ void __launch_bounds__(256)
charcnn(const int* __restrict__ x, const float* __restrict__ cemb,
        const _Float16* __restrict__ convwp /*[1408][64] padded f16*/,
        const float* __restrict__ convb, _Float16* __restrict__ feat /*[4096][1408] f16*/) {
  __shared__ _Float16 a_s[16 * CONVKP];
  int w = blockIdx.x;
  int s = w >> 5, bb = w & 31;                 // word n = s*32+b -> chars at x[b][s][:]
  const int* chars = x + (bb * SEQLEN + s) * WORDLEN;

  for (int i = threadIdx.x; i < 16 * CONVKP; i += 256) {
    int p = i >> 6, k = i & (CONVKP - 1);
    float v = 0.f;
    if (p < NPOS && k < 48) {                  // im2col col = e*3 + tap
      int e = k / 3, tap = k - e * 3;
      v = cemb[chars[p + tap] * CHAR_EMBC + e];
    }
    a_s[i] = (_Float16)v;
  }
  __syncthreads();

  int lane = threadIdx.x & 31;
  int wave = threadIdx.x >> 5;
  v16h a0 = loadA_h(a_s, CONVKP, 0, 0,  lane);
  v16h a1 = loadA_h(a_s, CONVKP, 0, 32, lane);

  int pbase = (lane >> 4) << 3;
  for (int ft = wave; ft < (NFILTP >> 4); ft += 8) {
    int n0 = ft << 4;
    v16h b0 = loadB_h(convwp, CONVKP, n0, 0,  lane);
    v16h b1 = loadB_h(convwp, CONVKP, n0, 32, lane);
    v8f acc = {};
    acc = __builtin_amdgcn_wmma_f32_16x16x32_f16(false, a0, false, b0, (short)0, acc, false, false);
    acc = __builtin_amdgcn_wmma_f32_16x16x32_f16(false, a1, false, b1, (short)0, acc, false, false);

    int f = n0 + (lane & 15);
    float bias = convb[f < NFILT ? f : NFILT - 1];   // branch-free clamp
    float mx = 0.f;                                  // max of relu >= 0
    #pragma unroll
    for (int r = 0; r < 8; ++r) {
      int p = pbase + r;
      if (p < NPOS) {
        float v = acc[r] + bias;
        mx = fmaxf(mx, v > 0.f ? v : 0.f);
      }
    }
    mx = fmaxf(mx, __shfl_xor(mx, 16, 32));          // combine pos 0-7 with 8-13
    if (lane < 16) feat[(long)w * NFILTP + f] = (_Float16)((f < NFILT) ? mx : 0.f);
  }
}

// ---------------- LSTM time scan: 1 block/direction, f16 h in LDS, c in registers ----------------
__global__ void __launch_bounds__(1024)
lstm_scan(const float* __restrict__ xw_fw, const float* __restrict__ xw_bw,
          const _Float16* __restrict__ fw_whh, const _Float16* __restrict__ bw_whh,
          float* __restrict__ outf, _Float16* __restrict__ out16) {
  __shared__ _Float16 h_s[BATCHN * HDIM];            // 16 KB
  int dir = blockIdx.x;
  const float*    xw  = dir ? xw_bw  : xw_fw;
  const _Float16* whh = dir ? bw_whh : fw_whh;
  int oofs = dir * HDIM;

  int lane = threadIdx.x & 31;
  int wave = threadIdx.x >> 5;
  int m0 = (wave >> 4) << 4;                         // batch tile (0 or 16)
  int j0 = (wave & 15) << 4;                         // hidden col tile
  int jcol  = j0 + (lane & 15);
  int mbase = m0 + ((lane >> 4) << 3);

  for (int i = threadIdx.x; i < BATCHN * HDIM; i += blockDim.x) h_s[i] = (_Float16)0.f;
  v8f c_reg = {};
  __syncthreads();

  for (int t = 0; t < SEQLEN; ++t) {
    v8f zero = {};
    v8f acc[4] = {zero, zero, zero, zero};           // i, f, g, o
    for (int k0 = 0; k0 < HDIM; k0 += 32) {
      v16h a = loadA_h(h_s, HDIM, m0, k0, lane);     // shared across 4 gate WMMAs
      #pragma unroll
      for (int g = 0; g < 4; ++g) {
        v16h b = loadB_h(whh, HDIM, g * HDIM + j0, k0, lane);
        acc[g] = __builtin_amdgcn_wmma_f32_16x16x32_f16(false, a, false, b, (short)0, acc[g], false, false);
      }
    }
    __syncthreads();                                 // all waves done reading h_s

    const float* xwt = xw + (long)t * BATCHN * (4 * HDIM);
    #pragma unroll
    for (int r = 0; r < 8; ++r) {
      int m = mbase + r;                             // batch index < 32
      const float* row = xwt + (long)m * (4 * HDIM);
      float gi = acc[0][r] + row[0 * HDIM + jcol];
      float gf = acc[1][r] + row[1 * HDIM + jcol];
      float gg = acc[2][r] + row[2 * HDIM + jcol];
      float go = acc[3][r] + row[3 * HDIM + jcol];
      gi = 1.f / (1.f + __expf(-gi));
      gf = 1.f / (1.f + __expf(-gf));
      gg = tanhf(gg);
      go = 1.f / (1.f + __expf(-go));
      float c = gf * c_reg[r] + gi * gg;
      c_reg[r] = c;
      float h = go * tanhf(c);
      long oi = ((long)t * BATCHN + m) * DDIM + oofs + jcol;
      h_s[m * HDIM + jcol] = (_Float16)h;
      outf[oi]  = h;
      out16[oi] = (_Float16)h;
    }
    __syncthreads();                                 // h visible before next step
  }
}

// ---------------- mean over batch dim (f32 in, f16 out for final GEMM) ----------------
__global__ void pool_mean(const float* __restrict__ lo, _Float16* __restrict__ pooled) {
  int idx = blockIdx.x * blockDim.x + threadIdx.x;
  if (idx >= SEQLEN * DDIM) return;
  int s = idx >> 9, d = idx & 511;
  float sum = 0.f;
  #pragma unroll 4
  for (int b = 0; b < BATCHN; ++b) sum += lo[((long)(s * BATCHN + b)) * DDIM + d];
  pooled[idx] = (_Float16)(sum * (1.f / 32.f));
}

extern "C" void kernel_launch(void* const* d_in, const int* in_sizes, int n_in,
                              void* d_out, int out_size, void* d_ws, size_t ws_size,
                              hipStream_t stream) {
  (void)in_sizes; (void)n_in; (void)out_size; (void)ws_size;
  const int*   x       = (const int*)  d_in[0];
  const float* cemb    = (const float*)d_in[1];
  const float* convw   = (const float*)d_in[2];
  const float* convb   = (const float*)d_in[3];
  const float* fcw_cnn = (const float*)d_in[4];
  const float* fcb_cnn = (const float*)d_in[5];
  const float* fw_wih  = (const float*)d_in[6];
  const float* fw_whh  = (const float*)d_in[7];
  const float* fw_bih  = (const float*)d_in[8];
  const float* fw_bhh  = (const float*)d_in[9];
  const float* bw_wih  = (const float*)d_in[10];
  const float* bw_whh  = (const float*)d_in[11];
  const float* bw_bih  = (const float*)d_in[12];
  const float* bw_bhh  = (const float*)d_in[13];
  const float* fcw     = (const float*)d_in[14];
  const float* fcb     = (const float*)d_in[15];

  float* outp       = (float*)d_out;
  float* logits     = outp;                                  // [128, 50000]
  float* lstm_final = outp + (size_t)SEQLEN * VOCABN;        // [128, 32, 512]

  char*  ws  = (char*)d_ws;
  size_t off = 0;
  auto alloc = [&](size_t bytes) -> void* {
    void* p = (void*)(ws + off);
    off += (bytes + 255) & ~(size_t)255;
    return p;
  };
  _Float16* convwp16 = (_Float16*)alloc((size_t)NFILTP * CONVKP * 2);
  _Float16* cnnfcp16 = (_Float16*)alloc((size_t)DDIM * NFILTP * 2);
  _Float16* wihf16   = (_Float16*)alloc((size_t)(4 * HDIM) * DDIM * 2);
  _Float16* wihb16   = (_Float16*)alloc((size_t)(4 * HDIM) * DDIM * 2);
  _Float16* whhf16   = (_Float16*)alloc((size_t)(4 * HDIM) * HDIM * 2);
  _Float16* whhb16   = (_Float16*)alloc((size_t)(4 * HDIM) * HDIM * 2);
  _Float16* fcw16    = (_Float16*)alloc((size_t)VOCABN * DDIM * 2);
  _Float16* feat16   = (_Float16*)alloc((size_t)NWORDS * NFILTP * 2);
  _Float16* w16a     = (_Float16*)alloc((size_t)NWORDS * DDIM * 2);
  _Float16* w16b     = (_Float16*)alloc((size_t)NWORDS * DDIM * 2);
  float*    xwf      = (float*)alloc((size_t)NWORDS * 4 * HDIM * sizeof(float));
  float*    xwb      = (float*)alloc((size_t)NWORDS * 4 * HDIM * sizeof(float));
  _Float16* pooled16 = (_Float16*)alloc((size_t)SEQLEN * DDIM * 2);

  auto pack = [&](const float* src, _Float16* dst, int N, int K, int Np, int Kp) {
    long e = (long)Np * Kp;
    pack_pad_h<<<(int)((e + 255) / 256), 256, 0, stream>>>(src, dst, N, K, Np, Kp);
  };

  // 0. one-time f32->f16 weight repacks (zero-padded where misaligned)
  pack(convw,   convwp16, NFILT, 48,    NFILTP, CONVKP);
  pack(fcw_cnn, cnnfcp16, DDIM,  NFILT, DDIM,   NFILTP);
  pack(fw_wih,  wihf16, 4 * HDIM, DDIM, 4 * HDIM, DDIM);
  pack(bw_wih,  wihb16, 4 * HDIM, DDIM, 4 * HDIM, DDIM);
  pack(fw_whh,  whhf16, 4 * HDIM, HDIM, 4 * HDIM, HDIM);
  pack(bw_whh,  whhb16, 4 * HDIM, HDIM, 4 * HDIM, HDIM);
  pack(fcw,     fcw16,  VOCABN, DDIM, VOCABN, DDIM);

  // 1. fused char-CNN -> feat16 [4096, 1408]
  charcnn<<<NWORDS, 256, 0, stream>>>(x, cemb, convwp16, convb, feat16);

  auto gemm = [&](const _Float16* A, const _Float16* W, const float* b1, const float* b2,
                  float* Cf, _Float16* Ch, int M, int N, int K) {
    long tasks  = (long)(M / 64) * (N / 16);
    int  blocks = (int)((tasks + 7) / 8);
    gemm_wt<<<blocks, 256, 0, stream>>>(A, W, b1, b2, Cf, Ch, M, N, K);
  };

  // 2. CNN projection -> words (f16) [4096, 512]
  gemm(feat16, cnnfcp16, fcb_cnn, nullptr, nullptr, w16a, NWORDS, DDIM, NFILTP);

  // 3. two biLSTM layers (weights reused each layer; batch-flip in ref cancels out)
  _Float16* cur16 = w16a;
  for (int layer = 0; layer < 2; ++layer) {
    _Float16* out16 = (cur16 == w16a) ? w16b : w16a;
    gemm(cur16, wihf16, fw_bih, fw_bhh, xwf, nullptr, NWORDS, 4 * HDIM, DDIM);
    gemm(cur16, wihb16, bw_bih, bw_bhh, xwb, nullptr, NWORDS, 4 * HDIM, DDIM);
    // f32 sink is the final lstm_output region both times (layer 2 overwrites layer 1)
    lstm_scan<<<2, 1024, 0, stream>>>(xwf, xwb, whhf16, whhb16, lstm_final, out16);
    cur16 = out16;
  }

  // 4. mean over batch, final vocab projection
  pool_mean<<<(SEQLEN * DDIM + 255) / 256, 256, 0, stream>>>(lstm_final, pooled16);
  gemm(pooled16, fcw16, fcb, nullptr, logits, nullptr, SEQLEN, VOCABN, DDIM);
}
